// MM_LSTM_Age_v51_70179765616726
// MI455X (gfx1250) — matrix operand order repaired and checked by reading, not compiled, not run on hardware
//
#include <hip/hip_runtime.h>
#include <math.h>

typedef __bf16 bf16;
typedef __attribute__((ext_vector_type(16))) __bf16 v16bf;
typedef __attribute__((ext_vector_type(8)))  __bf16 v8bf;
typedef __attribute__((ext_vector_type(8)))  float  v8f;

union AFrag { v16bf v; v8bf h[2]; };
union CFrag { v8f v; float f[8]; };

__constant__ float c_lsmax[7] = {-1.0516589f, -0.64791672f, -0.99871467f,
                                 -0.86651843f, -0.72868288f, -0.72231681f,
                                 -0.78230592f};

__device__ __forceinline__ v8f v8f_zero() {
  v8f z = {0.f,0.f,0.f,0.f,0.f,0.f,0.f,0.f};
  return z;
}
__device__ __forceinline__ float sigmf(float x) { return 1.0f / (1.0f + __expf(-x)); }

// A fragment (16x32 bf16, wave32): lane half=0 -> K {kb+0..7, kb+16..23},
// half=1 -> K {kb+8..15, kb+24..31}; p = row_base + kb.
__device__ __forceinline__ v16bf load_a16(const bf16* p, int half) {
  AFrag a;
  a.h[0] = *(const v8bf*)(p + half * 8);
  a.h[1] = *(const v8bf*)(p + 16 + half * 8);
  return a.v;
}
// B fragment (32x16 bf16): lane col = lane&15, K same as A: half=0 -> {kb+0..7, kb+16..23}, half=1 -> {kb+8..15, kb+24..31}.
// p = &Wt[col_j][kb] with Wt row-major [N][K] (pre-transposed weights).
__device__ __forceinline__ v16bf load_b16(const bf16* p, int half) {
  AFrag b;
  b.h[0] = *(const v8bf*)(p + half * 8);
  b.h[1] = *(const v8bf*)(p + 16 + half * 8);
  return b.v;
}
__device__ __forceinline__ v8f wmma_bf(v16bf a, v16bf b, v8f c) {
  return __builtin_amdgcn_wmma_f32_16x16x32_bf16(false, a, false, b, (short)0, c,
                                                 false, false);
}

// z = [x, h] @ Wc for the 4 gate column-tiles (n0, n0+256, n0+512, n0+768),
// one 16-row tile at rbase.  A from LDS, B from global bf16 Wc^T [1024][K].
// K loop split into x-part and h-part; unroll 2 bounds B-prefetch depth.
template <int KSTEPS, int SPLIT, int XSTR>
__device__ __forceinline__ void lstm_cols1(const bf16* __restrict__ sXr,
                                           const bf16* __restrict__ sHr,
                                           const bf16* __restrict__ wct, int K,
                                           int n0, int lq, int half, int rbase,
                                           v8f* acc) {
  const bf16* ar = sXr + (rbase + lq) * XSTR;
  const bf16* hrw = sHr + (rbase + lq) * 256;
  const bf16* wg0 = wct + (size_t)(n0 + lq) * K;
  const bf16* wg1 = wg0 + (size_t)256 * K;
  const bf16* wg2 = wg1 + (size_t)256 * K;
  const bf16* wg3 = wg2 + (size_t)256 * K;
  constexpr int KX = SPLIT / 32;
#pragma unroll 2
  for (int ks = 0; ks < KX; ++ks) {
    const int kb = ks * 32;
    v16bf a = load_a16(ar + kb, half);
    acc[0] = wmma_bf(a, load_b16(wg0 + kb, half), acc[0]);
    acc[1] = wmma_bf(a, load_b16(wg1 + kb, half), acc[1]);
    acc[2] = wmma_bf(a, load_b16(wg2 + kb, half), acc[2]);
    acc[3] = wmma_bf(a, load_b16(wg3 + kb, half), acc[3]);
  }
#pragma unroll 2
  for (int ks = KX; ks < KSTEPS; ++ks) {
    const int kb = ks * 32;
    v16bf a = load_a16(hrw + (kb - SPLIT), half);
    acc[0] = wmma_bf(a, load_b16(wg0 + kb, half), acc[0]);
    acc[1] = wmma_bf(a, load_b16(wg1 + kb, half), acc[1]);
    acc[2] = wmma_bf(a, load_b16(wg2 + kb, half), acc[2]);
    acc[3] = wmma_bf(a, load_b16(wg3 + kb, half), acc[3]);
  }
}

template <bool SIGACT>
__device__ __forceinline__ void lstm_update1(const v8f* acc,
                                             const float* __restrict__ bias,
                                             int col, float* cst, float* hout) {
  const float bi = bias[col], bff = bias[256 + col];
  const float bg = bias[512 + col], bo = bias[768 + col];
  CFrag zi, zf, zg, zo;
  zi.v = acc[0]; zf.v = acc[1]; zg.v = acc[2]; zo.v = acc[3];
#pragma unroll
  for (int r = 0; r < 8; ++r) {
    float iv = sigmf(zi.f[r] + bi);
    float fv = sigmf(zf.f[r] + bff);
    float gv = SIGACT ? sigmf(zg.f[r] + bg) : tanhf(zg.f[r] + bg);
    float ov = sigmf(zo.f[r] + bo);
    float cn = fv * cst[r] + iv * gv;
    cst[r] = cn;
    hout[r] = ov * (SIGACT ? sigmf(cn) : tanhf(cn));
  }
}

// ---------------- weight prep (f32 -> bf16, transpose/concat) ----------------

// Wc_eco^T [1024][288]: k<16 -> W[k][j]; 16..31 -> 0 (x pad); k>=32 -> U[k-32][j]
__global__ void prep_eco_w(const float* __restrict__ W, const float* __restrict__ U,
                           bf16* __restrict__ wct) {
  int idx = blockIdx.x * 256 + threadIdx.x;
  if (idx >= 1024 * 288) return;
  int j = idx / 288, k = idx % 288;
  float v = 0.f;
  if (k < 16) v = W[k * 1024 + j];
  else if (k >= 32) v = U[(k - 32) * 1024 + j];
  wct[idx] = (bf16)v;
}

// 21 matrices [(br,lstm)] of Wc^T [1024][512]: k<256 -> W[br][k][j]; else U[br][k-256][j]
__global__ void prep_br_w(const float* __restrict__ dW, const float* __restrict__ dU,
                          const float* __restrict__ aW, const float* __restrict__ aU,
                          const float* __restrict__ tW, const float* __restrict__ tU,
                          bf16* __restrict__ wc) {
  size_t idx = (size_t)blockIdx.x * 256 + threadIdx.x;
  if (idx >= (size_t)21 * 1024 * 512) return;
  int u = (int)(idx >> 19);  // /(1024*512)
  int br = u / 3, l = u % 3;
  int j = (int)((idx >> 9) & 1023);
  int k = (int)(idx & 511);
  const float* Wp = (l == 0) ? dW : (l == 1) ? aW : tW;
  const float* Up = (l == 0) ? dU : (l == 1) ? aU : tU;
  float v = (k < 256) ? Wp[((size_t)br * 256 + k) * 1024 + j]
                      : Up[((size_t)br * 256 + (k - 256)) * 1024 + j];
  wc[idx] = (bf16)v;
}

__global__ void prep_d_w(const float* __restrict__ d1W, const float* __restrict__ d2W,
                         bf16* __restrict__ d1t, bf16* __restrict__ d2t) {
  int idx = blockIdx.x * 256 + threadIdx.x;
  if (idx < 7 * 256 * 256) {
    int br = idx / 65536, rj = idx % 65536, j = rj / 256, k = rj % 256;
    d1t[idx] = (bf16)d1W[((size_t)br * 256 + k) * 256 + j];
  }
  if (idx < 7 * 64 * 256) {
    int br = idx / 16384, rj = idx % 16384, j = rj / 256, k = rj % 256;
    d2t[idx] = (bf16)d2W[((size_t)br * 256 + k) * 64 + j];
  }
}

// ---------------- eco LSTM scan: grid = 8 batch tiles, 512 thr (16 waves) ----
// Double-buffered h: new h written right after gate math (no hout-across-barrier).
// Wave w -> column tile w; 2 row-tile tasks (rt=0,1).

__global__ __launch_bounds__(512, 1) void eco_scan_kernel(
    const float* __restrict__ inputs, const bf16* __restrict__ wct,
    const float* __restrict__ bias, bf16* __restrict__ xecos) {
  __shared__ __align__(16) bf16 sX[32 * 32];
  __shared__ __align__(16) bf16 sH[2][32 * 256];
  const int tid = threadIdx.x;
  const int b0 = blockIdx.x * 32;
  for (int i = tid; i < 32 * 32; i += 512) sX[i] = (bf16)0.0f;
  for (int i = tid; i < 32 * 256; i += 512) sH[0][i] = (bf16)0.0f;
  const int lane = tid & 31, w = tid >> 5;
  const int lq = lane & 15, half = lane >> 4;
  const int n0 = (w & 15) * 16;
  float cst[16];
#pragma unroll
  for (int r = 0; r < 16; ++r) cst[r] = 0.f;
  __syncthreads();
  const int xrow = tid >> 4, xcol = tid & 15;
  for (int t = 0; t < 120; ++t) {
    const int cur = t & 1, nxt = cur ^ 1;
    sX[xrow * 32 + xcol] =
        (bf16)inputs[((size_t)(b0 + xrow) * 120 + t) * 38 + xcol];
    __syncthreads();
#pragma unroll
    for (int q = 0; q < 2; ++q) {
      const int rbase = q * 16;
      v8f acc[4];
#pragma unroll
      for (int i = 0; i < 4; ++i) acc[i] = v8f_zero();
      lstm_cols1<9, 32, 32>(sX, sH[cur], wct, 288, n0, lq, half, rbase, acc);
      float hq[8];
      lstm_update1<false>(acc, bias, n0 + lq, cst + q * 8, hq);
      const int col = n0 + lq;
#pragma unroll
      for (int r = 0; r < 8; ++r) {
        int row = rbase + r + 8 * half;
        bf16 hv = (bf16)hq[r];
        sH[nxt][row * 256 + col] = hv;
        xecos[((size_t)(b0 + row) * 120 + t) * 256 + col] = hv;
      }
    }
    __syncthreads();  // sX reads done (next xload) + sH[nxt] visible
  }
}

// ----- branch scan: data+att+ts fused per step; grid (8 tiles, 7 branches) ---
// 512 threads (16 waves).  Double-buffered h for all three LSTMs; 3 barriers/step.
// Phase 1: wave w -> lstm (w>>3), tasks (ct, rt): ct in {w&7, (w&7)+8}, rt in {0,1}.
// Phase 3 (ts): wave w -> column tile w, rt in {0,1}.

__global__ __launch_bounds__(512, 1) void branch_scan_kernel(
    const bf16* __restrict__ xecos, const bf16* __restrict__ wcall,
    const float* __restrict__ dB, const float* __restrict__ aB,
    const float* __restrict__ tB, bf16* __restrict__ hsel) {
  extern __shared__ char dsm_raw[];
  bf16* sm = (bf16*)dsm_raw;
  bf16* sX = sm;                                        // [32][256]
  bf16* sHD[2] = {sm + 1 * 8192, sm + 2 * 8192};        // data h (cur/nxt)
  bf16* sHA[2] = {sm + 3 * 8192, sm + 4 * 8192};        // att h
  bf16* sHT[2] = {sm + 5 * 8192, sm + 6 * 8192};        // ts h
  bf16* sPR = sm + 7 * 8192;                            // h_data * h_att
  const int tid = threadIdx.x;  // 512
  const int b0 = blockIdx.x * 32;
  const int br = blockIdx.y;
  const bf16* wcD = wcall + (size_t)(br * 3 + 0) * 1024 * 512;
  const bf16* wcA = wcall + (size_t)(br * 3 + 1) * 1024 * 512;
  const bf16* wcT = wcall + (size_t)(br * 3 + 2) * 1024 * 512;
  const float* bT = tB + br * 1024;
  for (int i = tid; i < 8192; i += 512) {
    sHD[0][i] = (bf16)0.f; sHA[0][i] = (bf16)0.f; sHT[0][i] = (bf16)0.f;
  }
  const int lane = tid & 31, w = tid >> 5;  // 16 waves
  const int lq = lane & 15, half = lane >> 4;
  const int isAtt = w >> 3;  // waves 0-7: data, 8-15: att
  const int ct0 = w & 7;
  const bf16* wc1 = isAtt ? wcA : wcD;
  const float* b1 = isAtt ? (aB + br * 1024) : (dB + br * 1024);
  float cst[32];   // phase-1: 4 tasks x 8
  float cstT[16];  // ts: 2 tasks x 8
#pragma unroll
  for (int r = 0; r < 32; ++r) cst[r] = 0.f;
#pragma unroll
  for (int r = 0; r < 16; ++r) cstT[r] = 0.f;
  const int n0T = w * 16;
  __syncthreads();
  for (int t = 0; t < 120; ++t) {
    const int cur = t & 1, nxt = cur ^ 1;
    bf16* h1c = isAtt ? sHA[cur] : sHD[cur];
    bf16* h1n = isAtt ? sHA[nxt] : sHD[nxt];
    // load x_t tile [32][256] bf16
    for (int i = tid; i < 1024; i += 512) {
      int row = i >> 5, c8 = i & 31;
      ((v8bf*)sX)[i] =
          *(const v8bf*)(xecos + ((size_t)(b0 + row) * 120 + t) * 256 + c8 * 8);
    }
    __syncthreads();  // B1: x ready, prev-step h[cur] stable
    // phase 1: data & att LSTMs (K = 512: [x | h]); 4 tasks, rt inner for B reuse
#pragma unroll
    for (int q = 0; q < 4; ++q) {
      const int ct = ct0 + (q >> 1) * 8;
      const int rbase = (q & 1) * 16;
      const int n0 = ct * 16;
      v8f acc[4];
#pragma unroll
      for (int i = 0; i < 4; ++i) acc[i] = v8f_zero();
      lstm_cols1<16, 256, 256>(sX, h1c, wc1, 512, n0, lq, half, rbase, acc);
      float hq[8];
      if (isAtt) lstm_update1<true>(acc, b1, n0 + lq, cst + q * 8, hq);
      else       lstm_update1<false>(acc, b1, n0 + lq, cst + q * 8, hq);
      const int col = n0 + lq;
#pragma unroll
      for (int r = 0; r < 8; ++r)
        h1n[(rbase + r + 8 * half) * 256 + col] = (bf16)hq[r];
    }
    __syncthreads();  // B2: h[nxt] complete; sX reads done
    {
      const bf16* hdn = sHD[nxt];
      const bf16* han = sHA[nxt];
      for (int i = tid; i < 8192; i += 512)
        sPR[i] = (bf16)((float)hdn[i] * (float)han[i]);
    }
    __syncthreads();  // B3: prod ready
    // phase 3: ts LSTM (input = prod); 2 row-tile tasks
    {
      const bool samp = (t % 12) == 11;
      const int s = (t - 11) / 12;
#pragma unroll
      for (int q = 0; q < 2; ++q) {
        const int rbase = q * 16;
        v8f acc[4];
#pragma unroll
        for (int i = 0; i < 4; ++i) acc[i] = v8f_zero();
        lstm_cols1<16, 256, 256>(sPR, sHT[cur], wcT, 512, n0T, lq, half, rbase,
                                 acc);
        float hq[8];
        lstm_update1<false>(acc, bT, n0T + lq, cstT + q * 8, hq);
        const int col = n0T + lq;
#pragma unroll
        for (int r = 0; r < 8; ++r) {
          int row = rbase + r + 8 * half;
          bf16 hv = (bf16)hq[r];
          sHT[nxt][row * 256 + col] = hv;
          if (samp)
            hsel[(((size_t)br * 10 + s) * 256 + (b0 + row)) * 256 + col] = hv;
        }
      }
    }
    // loop: next xload only touches sX; B1 orders sHT[nxt] before next ts reads
  }
}

// ---------------- MLP head: d1(256) relu, d2(64) relu, df(67->1) -------------

template <int KSTEPS>
__device__ __forceinline__ v8f dense_tile(const bf16* __restrict__ sA,
                                          const bf16* __restrict__ wt, int K,
                                          int j0, int lq, int half, int rbase) {
  v8f acc = v8f_zero();
  const bf16* ar = sA + (rbase + lq) * 256;
#pragma unroll
  for (int ks = 0; ks < KSTEPS; ++ks) {
    const int kb = ks * 32;
    v16bf a = load_a16(ar + kb, half);
    v16bf b = load_b16(wt + (size_t)(j0 + lq) * K + kb, half);
    acc = wmma_bf(a, b, acc);
  }
  return acc;
}

__global__ __launch_bounds__(256, 1) void head_kernel(
    const bf16* __restrict__ hsel, const bf16* __restrict__ d1t,
    const float* __restrict__ d1b, const bf16* __restrict__ d2t,
    const float* __restrict__ d2b, const float* __restrict__ dfW,
    const float* __restrict__ dfb, const float* __restrict__ inputs,
    float* __restrict__ out) {
  __shared__ __align__(16) bf16 sA[32 * 256];
  __shared__ __align__(16) bf16 sB1[32 * 256];
  __shared__ __align__(16) float sC2[32 * 64];
  const int tid = threadIdx.x;  // 256
  const int br = blockIdx.y;
  const int r0 = blockIdx.x * 32;  // flat row = s*256 + b
  const bf16* src = hsel + ((size_t)br * 2560 + r0) * 256;
  for (int i = tid; i < 1024; i += 256) ((v8bf*)sA)[i] = ((const v8bf*)src)[i];
  __syncthreads();
  const int lane = tid & 31, w = tid >> 5;
  const int lq = lane & 15, half = lane >> 4;
  const bf16* w1 = d1t + (size_t)br * 256 * 256;
#pragma unroll
  for (int q = 0; q < 4; ++q) {  // d1: 32 tiles / 8 waves
    int tg = w + 8 * q, rt = tg >> 4, ct = tg & 15;
    int rbase = rt * 16, j0 = ct * 16;
    v8f acc = dense_tile<8>(sA, w1, 256, j0, lq, half, rbase);
    CFrag c; c.v = acc;
    int col = j0 + lq;
    float bb = d1b[br * 256 + col];
#pragma unroll
    for (int r = 0; r < 8; ++r) {
      float v = fmaxf(c.f[r] + bb, 0.f);
      sB1[(rbase + r + 8 * half) * 256 + col] = (bf16)v;
    }
  }
  __syncthreads();
  {  // d2: 8 tiles / 8 waves
    int rt = w >> 2, ct = w & 3, rbase = rt * 16, j0 = ct * 16;
    v8f acc = dense_tile<8>(sB1, d2t + (size_t)br * 64 * 256, 256, j0, lq, half, rbase);
    CFrag c; c.v = acc;
    int col = j0 + lq;
    float bb = d2b[br * 64 + col];
#pragma unroll
    for (int r = 0; r < 8; ++r) {
      float v = fmaxf(c.f[r] + bb, 0.f);
      sC2[(rbase + r + 8 * half) * 64 + col] = v;
    }
  }
  __syncthreads();
  if (tid < 32) {
    int fr = r0 + tid, s = fr >> 8, b = fr & 255;
    const float* wf = dfW + br * 67;
    float a = dfb[br];
    for (int k = 0; k < 64; ++k) a += sC2[tid * 64 + k] * wf[k];
    const float* row11 = inputs + ((size_t)b * 120 + 11) * 38;
    a += row11[23] * wf[64] + row11[24 + br] * wf[65] + row11[31 + br] * wf[66];
    a += row11[16 + br];  // init
    out[((size_t)b * 10 + s) * 7 + br] = fmaxf(a, c_lsmax[br]);
  }
}

// ---------------- workspace layout & launch ----------------------------------

static constexpr size_t AL(size_t x) { return (x + 255) & ~(size_t)255; }
static constexpr size_t OFF_WCECO = 0;
static constexpr size_t OFF_WCBR = AL(OFF_WCECO + (size_t)1024 * 288 * 2);
static constexpr size_t OFF_D1T = AL(OFF_WCBR + (size_t)21 * 1024 * 512 * 2);
static constexpr size_t OFF_D2T = AL(OFF_D1T + (size_t)7 * 256 * 256 * 2);
static constexpr size_t OFF_XE = AL(OFF_D2T + (size_t)7 * 64 * 256 * 2);
static constexpr size_t OFF_HSEL = AL(OFF_XE + (size_t)256 * 120 * 256 * 2);

extern "C" void kernel_launch(void* const* d_in, const int* in_sizes, int n_in,
                              void* d_out, int out_size, void* d_ws,
                              size_t ws_size, hipStream_t stream) {
  (void)in_sizes; (void)n_in; (void)out_size; (void)ws_size;
  const float* inputs = (const float*)d_in[0];
  const float* ecoW = (const float*)d_in[1];
  const float* ecoU = (const float*)d_in[2];
  const float* ecob = (const float*)d_in[3];
  const float* dW = (const float*)d_in[4];
  const float* dU = (const float*)d_in[5];
  const float* db_ = (const float*)d_in[6];
  const float* aW = (const float*)d_in[7];
  const float* aU = (const float*)d_in[8];
  const float* ab_ = (const float*)d_in[9];
  const float* tW = (const float*)d_in[10];
  const float* tU = (const float*)d_in[11];
  const float* tb_ = (const float*)d_in[12];
  const float* d1W = (const float*)d_in[13];
  const float* d1b = (const float*)d_in[14];
  const float* d2W = (const float*)d_in[15];
  const float* d2b = (const float*)d_in[16];
  const float* dfW = (const float*)d_in[17];
  const float* dfb = (const float*)d_in[18];

  char* ws = (char*)d_ws;
  bf16* wceco = (bf16*)(ws + OFF_WCECO);
  bf16* wcbr = (bf16*)(ws + OFF_WCBR);
  bf16* d1t = (bf16*)(ws + OFF_D1T);
  bf16* d2t = (bf16*)(ws + OFF_D2T);
  bf16* xe = (bf16*)(ws + OFF_XE);
  bf16* hsel = (bf16*)(ws + OFF_HSEL);

  prep_eco_w<<<(1024 * 288 + 255) / 256, 256, 0, stream>>>(ecoW, ecoU, wceco);
  prep_br_w<<<(int)(((size_t)21 * 1024 * 512 + 255) / 256), 256, 0, stream>>>(
      dW, dU, aW, aU, tW, tU, wcbr);
  prep_d_w<<<(7 * 256 * 256 + 255) / 256, 256, 0, stream>>>(d1W, d2W, d1t, d2t);

  eco_scan_kernel<<<8, 512, 0, stream>>>(inputs, wceco, ecob, xe);
  branch_scan_kernel<<<dim3(8, 7), 512, 8 * 8192 * sizeof(bf16), stream>>>(
      xe, wcbr, db_, ab_, tb_, hsel);
  head_kernel<<<dim3(80, 7), 256, 0, stream>>>(hsel, d1t, d1b, d2t, d2b, dfW,
                                               dfb, inputs, (float*)d_out);
}
